// VectorQuantizer_81793357185304
// MI455X (gfx1250) — matrix operand order repaired and checked by reading, not compiled, not run on hardware
//
#include <hip/hip_runtime.h>
#include <hip/hip_bf16.h>
#include <cfloat>
#include <climits>

typedef _Float16 half8  __attribute__((ext_vector_type(8)));
typedef _Float16 half16 __attribute__((ext_vector_type(16)));
typedef float    float8 __attribute__((ext_vector_type(8)));

static constexpr int D  = 256;
static constexpr int K  = 1024;
static constexpr int BN = 64 * 512;   // 32768 rows

// workspace layout (float indices into d_ws)
static constexpr int OFF_NV  = 512;     // 1   float : max(n_valid,1)
static constexpr int OFF_MU  = 768;     // 256 float : per-dim mean
static constexpr int OFF_SC  = 1024;    // 256 float : gamma*rsqrt(var+eps)
static constexpr int OFF_E2  = 1536;    // 1024 float: ||e_col||^2
static constexpr int OFF_CNT = 2560;    // 1024 int  : code counts
static constexpr int OFF_IDX = 3584;    // 32768 int : argmin per row
static constexpr int OFF_RL  = 36352;   // 32768 float: per-row loss
static constexpr int OFF_EH  = 69120;   // 262144 half: swizzled f16 codebook (B-frag layout)
static constexpr int OFF_PS  = 200192;  // 128*256 float: partial sums
static constexpr int OFF_PS2 = 232960;  // 128*256 float: partial sumsq
static constexpr int OFF_PN  = 265728;  // 128 float: partial valid counts
static constexpr int OFF_ET  = 265984;  // 262144 float: e transposed [K][D] for coalesced gather

// output layout (floats)
static constexpr size_t OUT_DL  = 4194304;
static constexpr size_t OUT_CL  = 4194305;
static constexpr size_t OUT_IDX = 4194306;
static constexpr size_t OUT_PP  = 4227074;

// ---- stage 1: masked per-dim partial sums (deterministic, no FP atomics) ----
__global__ __launch_bounds__(256) void vq_reduce(const float* __restrict__ x,
                                                 const int* __restrict__ mask,
                                                 float* __restrict__ W) {
  const int t = threadIdx.x;               // dim
  const int b = blockIdx.x;                // 128 blocks, 256 rows each
  const int r0 = b * 256;
  float s = 0.f, s2 = 0.f, nl = 0.f;
  for (int r = 0; r < 256; ++r) {
    const int row = r0 + r;
    const float m  = (float)mask[row];
    const float xv = x[(size_t)row * D + t];
    const float mx = m * xv;
    s += mx; s2 += mx * xv; nl += m;
  }
  W[OFF_PS  + b * 256 + t] = s;
  W[OFF_PS2 + b * 256 + t] = s2;
  if (t == 0) W[OFF_PN + b] = nl;
}

// ---- stage 2: finalize BN statistics ----
__global__ __launch_bounds__(256) void vq_finalize(float* __restrict__ W,
                                                   const float* __restrict__ gamma) {
  const int t = threadIdx.x;
  float s = 0.f, s2 = 0.f;
  for (int b = 0; b < 128; ++b) {
    s  += W[OFF_PS  + b * 256 + t];
    s2 += W[OFF_PS2 + b * 256 + t];
  }
  if (t == 0) {
    float n = 0.f;
    for (int b = 0; b < 128; ++b) n += W[OFF_PN + b];
    W[OFF_NV] = fmaxf(n, 1.0f);
  }
  __syncthreads();
  const float n  = W[OFF_NV];
  const float mu = s / n;
  const float var = fmaxf(s2 / n - mu * mu, 0.0f);
  W[OFF_MU + t] = mu;
  W[OFF_SC + t] = gamma[t] * rsqrtf(var + 1e-5f);
}

// ---- per-code squared norms + transpose to eT[K][D] for coalesced gather ----
__global__ __launch_bounds__(256) void vq_e2t(const float* __restrict__ e,
                                              float* __restrict__ W) {
  const int k = blockIdx.x * 256 + threadIdx.x;
  float s = 0.f;
  for (int d = 0; d < D; ++d) {
    const float v = e[(size_t)d * K + k];
    W[OFF_ET + (size_t)k * D + d] = v;
    s += v * v;
  }
  W[OFF_E2 + k] = s;
}

// ---- pre-swizzle codebook into f16 B-fragment layout ----
// B 32x16 f16 fragment: lane L holds column n=L&15; K = chunk*32 + (L>>4)*16 + j (j=0..15)
__global__ __launch_bounds__(256) void vq_swz(const float* __restrict__ e,
                                              _Float16* __restrict__ eh) {
  const int tid   = blockIdx.x * 256 + threadIdx.x;   // 0..16383 == (ct*8+chunk)*32+lane
  const int lane  = tid & 31;
  const int chunk = (tid >> 5) & 7;
  const int ct    = tid >> 8;
  const int c  = ct * 16 + (lane & 15);
  const int kb = chunk * 32 + ((lane >> 4) << 4);
  half16 v;
#pragma unroll
  for (int j = 0; j < 16; ++j) v[j] = (_Float16)e[(size_t)(kb + j) * K + c];
  *(half16*)&eh[(size_t)tid * 16] = v;
}

// ---- fused BN + distance GEMM (WMMA f16) + argmin ----
__global__ __launch_bounds__(256) void vq_main(const float* __restrict__ x,
                                               const float* __restrict__ W,
                                               const float* __restrict__ beta,
                                               const _Float16* __restrict__ eh,
                                               int* __restrict__ idxArr) {
  __shared__ __align__(32) _Float16 xbs[64 * 256];   // 32 KB: 64 rows of BN'd x in f16
  __shared__ float redv[64 * 8];
  __shared__ int   redi[64 * 8];
  const int t    = threadIdx.x;
  const int lane = t & 31;
  const int w    = t >> 5;                 // 8 waves
  const int r0   = blockIdx.x * 64;

  { // phase 1: BatchNorm 64 rows into LDS (thread t owns dim t)
    const float mud = W[OFF_MU + t];
    const float sc  = W[OFF_SC + t];
    const float bb  = beta[t];
    for (int r = 0; r < 64; ++r) {
      const float xv = x[(size_t)(r0 + r) * D + t];
      xbs[r * 256 + t] = (_Float16)((xv - mud) * sc + bb);
    }
  }
  __syncthreads();

  const int nsub = lane & 15;     // C/D column within tile, A row within tile
  const int ksel = lane >> 4;

  // preload all A fragments for this wave's 4 row-tiles (contiguous k order, same as B):
  // lane<16: K = kb..kb+15 (halves 0-15); lane>=16: kb=+16
  half16 af[4][8];
#pragma unroll
  for (int chunk = 0; chunk < 8; ++chunk) {
    const int kb = chunk * 32 + ksel * 16;
#pragma unroll
    for (int rt = 0; rt < 4; ++rt) {
      const int row = rt * 16 + nsub;
      const half8 lo = *(const half8*)&xbs[row * 256 + kb];
      const half8 hi = *(const half8*)&xbs[row * 256 + kb + 8];
      af[rt][chunk] = __builtin_shufflevector(lo, hi,
          0,1,2,3,4,5,6,7,8,9,10,11,12,13,14,15);
    }
  }

  float minv[4][8];
  int   mini[4][8];
#pragma unroll
  for (int rt = 0; rt < 4; ++rt)
#pragma unroll
    for (int i = 0; i < 8; ++i) { minv[rt][i] = FLT_MAX; mini[rt][i] = INT_MAX; }

  for (int cti = 0; cti < 8; ++cti) {          // 8 column tiles per wave
    const int ct = cti * 8 + w;                // 16 columns each
    // per-ct B region: eh + ct*4096 halves (8KB contiguous), chunk stride 512 halves
    const _Float16* bbase = eh + (size_t)ct * 4096 + (size_t)lane * 16;

    // warm next tile's 8KB B region into WGP$ (scope 0 = WGP: fills all cache levels)
    if (cti < 7) {
      const _Float16* np = eh + (size_t)(ct + 8) * 4096 + (size_t)lane * 128;
      asm volatile("global_prefetch_b8 %0, off\n\t"
                   "global_prefetch_b8 %0, off offset:128"
                   :: "v"(np) : "memory");
    }

    float8 acc[4] = {};
    half16 bcur = *(const half16*)bbase;       // chunk 0
#pragma unroll
    for (int chunk = 0; chunk < 8; ++chunk) {  // D=256 in chunks of 32
      half16 bnext;
      if (chunk < 7)                           // issue next chunk's load before consuming
        bnext = *(const half16*)(bbase + (size_t)(chunk + 1) * 512);
      // forbid the bnext load from sinking below / WMMAs from hoisting above:
      __builtin_amdgcn_sched_barrier(0);
#pragma unroll
      for (int rt = 0; rt < 4; ++rt) {         // 4 row tiles (64 rows)
        acc[rt] = __builtin_amdgcn_wmma_f32_16x16x32_f16(
            false, af[rt][chunk], false, bcur, (short)0, acc[rt], false, false);
      }
      if (chunk < 7) bcur = bnext;
    }

    const int   c   = ct * 16 + nsub;
    const float e2c = W[OFF_E2 + c];
#pragma unroll
    for (int rt = 0; rt < 4; ++rt)
#pragma unroll
      for (int i = 0; i < 8; ++i) {
        const float dv = e2c - 2.0f * acc[rt][i];   // row-constant ||xb||^2 dropped
        if (dv < minv[rt][i] || (dv == minv[rt][i] && c < mini[rt][i])) {
          minv[rt][i] = dv; mini[rt][i] = c;
        }
      }
  }

  // cross-lane argmin within each 16-lane half (cols live across lanes)
#pragma unroll
  for (int rt = 0; rt < 4; ++rt)
#pragma unroll
    for (int i = 0; i < 8; ++i) {
      float v = minv[rt][i]; int id = mini[rt][i];
#pragma unroll
      for (int mm = 8; mm >= 1; mm >>= 1) {
        const float ov = __shfl_xor(v, mm, 16);
        const int   oi = __shfl_xor(id, mm, 16);
        if (ov < v || (ov == v && oi < id)) { v = ov; id = oi; }
      }
      if (nsub == 0) {
        const int row = rt * 16 + i + (ksel << 3);
        redv[row * 8 + w] = v;
        redi[row * 8 + w] = id;
      }
    }
  __syncthreads();
  if (t < 64) {   // combine the 8 waves' column ranges
    float bv = FLT_MAX; int bi = INT_MAX;
    for (int ww = 0; ww < 8; ++ww) {
      const float v = redv[t * 8 + ww]; const int id = redi[t * 8 + ww];
      if (v < bv || (v == bv && id < bi)) { bv = v; bi = id; }
    }
    idxArr[r0 + t] = bi;
  }
}

// ---- gather / outputs / per-row loss (coalesced gather from eT) ----
__global__ __launch_bounds__(256) void vq_out(const float* __restrict__ x,
                                              const int* __restrict__ mask,
                                              float* __restrict__ W,
                                              const float* __restrict__ beta,
                                              const int* __restrict__ idxArr,
                                              float* __restrict__ out,
                                              int* __restrict__ counts) {
  __shared__ float sred[256];
  const int row = blockIdx.x;
  const int t   = threadIdx.x;
  const int m   = mask[row];
  const int id  = idxArr[row];
  const float q  = W[OFF_ET + (size_t)id * D + t];
  const float xv = x[(size_t)row * D + t];
  const float xb = (xv - W[OFF_MU + t]) * W[OFF_SC + t] + beta[t];
  const float df = xb - q;
  sred[t] = m ? df * df : 0.0f;
  out[(size_t)row * D + t] = m ? q : 0.0f;
  __syncthreads();
  for (int s = 128; s > 0; s >>= 1) { if (t < s) sred[t] += sred[t + s]; __syncthreads(); }
  if (t == 0) {
    W[OFF_RL + row] = sred[0];
    out[OUT_IDX + row] = m ? (float)id : -1.0f;
    if (m) atomicAdd(&counts[id], 1);
  }
}

// ---- scalars: losses + perplexity (fixed-order reductions) ----
__global__ __launch_bounds__(256) void vq_scalars(const float* __restrict__ W,
                                                  const int* __restrict__ counts,
                                                  float* __restrict__ out) {
  __shared__ float sred[256];
  const int t = threadIdx.x;
  const float n = W[OFF_NV];
  float s = 0.f;
  for (int i = t; i < BN; i += 256) s += W[OFF_RL + i];
  sred[t] = s; __syncthreads();
  for (int k = 128; k > 0; k >>= 1) { if (t < k) sred[t] += sred[t + k]; __syncthreads(); }
  const float loss = sred[0] / (n * (float)D);
  __syncthreads();
  float h = 0.f;
  for (int k = t; k < K; k += 256) {
    const float p = (float)counts[k] / n;
    h += p * logf(p + 1e-10f);
  }
  sred[t] = h; __syncthreads();
  for (int k = 128; k > 0; k >>= 1) { if (t < k) sred[t] += sred[t + k]; __syncthreads(); }
  if (t == 0) {
    out[OUT_DL] = loss;
    out[OUT_CL] = loss;
    out[OUT_PP] = expf(-sred[0]);
  }
}

extern "C" void kernel_launch(void* const* d_in, const int* in_sizes, int n_in,
                              void* d_out, int out_size, void* d_ws, size_t ws_size,
                              hipStream_t stream) {
  const float* x     = (const float*)d_in[0];
  const int*   mask  = (const int*)d_in[1];
  const float* e     = (const float*)d_in[2];
  const float* gamma = (const float*)d_in[3];
  const float* beta  = (const float*)d_in[4];
  float* out = (float*)d_out;
  float* W   = (float*)d_ws;
  _Float16* eh  = (_Float16*)(W + OFF_EH);
  int* counts   = (int*)(W + OFF_CNT);
  int* idxArr   = (int*)(W + OFF_IDX);

  hipMemsetAsync((char*)d_ws + (size_t)OFF_CNT * sizeof(float), 0,
                 (size_t)K * sizeof(int), stream);

  vq_reduce  <<<128,   256, 0, stream>>>(x, mask, W);
  vq_finalize<<<1,     256, 0, stream>>>(W, gamma);
  vq_e2t     <<<4,     256, 0, stream>>>(e, W);
  vq_swz     <<<64,    256, 0, stream>>>(e, eh);
  vq_main    <<<BN/64, 256, 0, stream>>>(x, W, beta, eh, idxArr);
  vq_out     <<<BN,    256, 0, stream>>>(x, mask, W, beta, idxArr, out, counts);
  vq_scalars <<<1,     256, 0, stream>>>(W, counts, out);
}
